// GraphAttentiveLinear_63376537420069
// MI455X (gfx1250) — matrix operand order repaired and checked by reading, not compiled, not run on hardware
//
#include <hip/hip_runtime.h>
#include <hip/hip_bf16.h>

// GraphAttentiveLinear fused kernel for MI455X (gfx1250, wave32, WMMA).
//
// Pipeline per 32-sample tile (one 256-thread block = 8 waves):
//   v -> Gram(32x32) -> sign/eps clamp -> LN1 -> LReLU -> x@W1^T (bf16 WMMA)
//     -> LN2 -> LReLU -> x@W2^T + b2 (bf16 WMMA) -> softmax(32) -> attn @ v
//
// TILE_M=32: each B (weight) fragment feeds TWO wmma (row-halves 0-15 / 16-31),
// halving L2 weight traffic vs TILE_M=16 (~34 FLOP per L2 byte).
// Weights are pre-converted f32 -> bf16 (RNE) into d_ws (needs 8 MiB).

#define C_DIM 32
#define NI    1024
#define TILE_M 32
#define LN_EPS 1e-5f
#define SLOPE  0.01f

typedef __attribute__((ext_vector_type(16))) __bf16 v16bf;
typedef __attribute__((ext_vector_type(8)))  float  v8f;

union ABfrag { v16bf v; uint4 q[2]; };

__device__ __forceinline__ unsigned short f2bf(float x) {
  union { float f; unsigned int u; } t; t.f = x;
  unsigned int r = t.u + 0x7fffu + ((t.u >> 16) & 1u);  // round-to-nearest-even
  return (unsigned short)(r >> 16);
}

__global__ void convert_f32_to_bf16(const float* __restrict__ src,
                                    unsigned short* __restrict__ dst, int n) {
  int i = blockIdx.x * blockDim.x + threadIdx.x;
  if (i < n) dst[i] = f2bf(src[i]);
}

// LayerNorm (biased var) + LeakyReLU + bf16 convert. Each wave handles 4 samples.
__device__ __forceinline__ void ln_lrelu_stage(const float* Xf32, unsigned short* Xb,
                                               const float* __restrict__ g,
                                               const float* __restrict__ b,
                                               int wid, int lane) {
  for (int si = 0; si < 4; ++si) {
    int s = wid * 4 + si;
    const float* row = Xf32 + s * NI;
    float sum = 0.f, sq = 0.f;
    for (int i = lane; i < NI; i += 32) { float x = row[i]; sum += x; sq += x * x; }
#pragma unroll
    for (int off = 16; off > 0; off >>= 1) {
      sum += __shfl_xor(sum, off, 32);
      sq  += __shfl_xor(sq,  off, 32);
    }
    float mean = sum * (1.f / NI);
    float var  = sq * (1.f / NI) - mean * mean;
    float rstd = rsqrtf(var + LN_EPS);
    unsigned short* orow = Xb + s * NI;
    for (int i = lane; i < NI; i += 32) {
      float y = (row[i] - mean) * rstd * g[i] + b[i];
      y = (y >= 0.f) ? y : SLOPE * y;
      orow[i] = f2bf(y);
    }
  }
}

// H[32,1024] = X[32,1024](bf16,LDS) x W^T (W is [out=1024, in=1024] bf16, row-major).
// Wave `wid` owns output columns [wid*128, wid*128+128) = 8 tile-columns of 16,
// each spanning two 16x16 WMMA row-halves (rows 0-15 and 16-31).
// Fragment packing per CDNA5 ISA 7.12.2 (16-bit A 16x32, B 32x16, 32-bit C/D 16x16).
__device__ __forceinline__ void gemm_1024(const unsigned short* Xb, float* Xf32,
                                          const unsigned short* __restrict__ W,
                                          const float* __restrict__ bias,
                                          int wid, int l15, int hf) {
  v8f acc0[8], acc1[8];
  v8f zero = {0.f, 0.f, 0.f, 0.f, 0.f, 0.f, 0.f, 0.f};
#pragma unroll
  for (int t = 0; t < 8; ++t) { acc0[t] = zero; acc1[t] = zero; }

  // A: lane L -> row M=L&15; VGPR j<4 holds K=8*hf+2j..+1, j>=4 holds 16+8*hf+...
  //    => contiguous bf16 runs [K0+8*hf, +8) and [K0+16+8*hf, +8).
  const unsigned short* arow0 = Xb + l15 * NI + 8 * hf;           // rows 0..15
  const unsigned short* arow1 = Xb + (16 + l15) * NI + 8 * hf;    // rows 16..31
  // B: lane L -> col N=L&15; same K layout as A => runs [K0+8*hf, +8) and [K0+16+8*hf, +8).
  const unsigned short* wbase = W + (size_t)(wid * 128 + l15) * NI + 8 * hf;

  for (int k = 0; k < 32; ++k) {
    const int K0 = k * 32;
    ABfrag fa0, fa1;
    fa0.q[0] = *(const uint4*)(arow0 + K0);
    fa0.q[1] = *(const uint4*)(arow0 + K0 + 16);
    fa1.q[0] = *(const uint4*)(arow1 + K0);
    fa1.q[1] = *(const uint4*)(arow1 + K0 + 16);
#pragma unroll
    for (int t = 0; t < 8; ++t) {
      const unsigned short* wrow = wbase + (size_t)t * 16 * NI + K0;
      ABfrag fb;
      fb.q[0] = *(const uint4*)(wrow);
      fb.q[1] = *(const uint4*)(wrow + 16);
      acc0[t] = __builtin_amdgcn_wmma_f32_16x16x32_bf16(
          false, fa0.v, false, fb.v, (short)0, acc0[t], false, false);
      acc1[t] = __builtin_amdgcn_wmma_f32_16x16x32_bf16(
          false, fa1.v, false, fb.v, (short)0, acc1[t], false, false);
    }
  }
  // C/D layout: VGPR r, lanes 0-15: (M=r, N=lane); lanes 16-31: (M=r+8, N=lane-16)
#pragma unroll
  for (int t = 0; t < 8; ++t) {
    int col = wid * 128 + t * 16 + l15;
    float badd = bias ? bias[col] : 0.f;
#pragma unroll
    for (int r = 0; r < 8; ++r) {
      Xf32[(r + 8 * hf) * NI + col]      = acc0[t][r] + badd;
      Xf32[(16 + r + 8 * hf) * NI + col] = acc1[t][r] + badd;
    }
  }
}

__global__ __launch_bounds__(256)
void gal_kernel(const float* __restrict__ f,            // [N, 32, m]
                const unsigned short* __restrict__ W1,  // bf16 [1024,1024]
                const unsigned short* __restrict__ W2,  // bf16 [1024,1024]
                const float* __restrict__ g1, const float* __restrict__ b1,
                const float* __restrict__ g2, const float* __restrict__ b2,
                const float* __restrict__ bias2,        // [1024]
                float* __restrict__ out,                // [N, 32, m]
                int m, int Ntot) {
  extern __shared__ __align__(16) char smem[];
  float*          Xf32 = (float*)smem;                               // 128 KB
  unsigned short* Xb   = (unsigned short*)(smem + TILE_M * NI * 4);  //  64 KB
  float*          vbuf = (float*)(smem + TILE_M * NI * 6);           //  12 KB

  const int tid  = threadIdx.x;
  const int wid  = tid >> 5;
  const int lane = tid & 31;
  const int l15  = lane & 15;
  const int hf   = lane >> 4;
  const int s0   = blockIdx.x * TILE_M;

  // ---- Stage 0: load v tile into LDS ----
  {
    int total = TILE_M * C_DIM * m;
    int avail = (Ntot - s0) * C_DIM * m;
    const float* src = f + (size_t)s0 * C_DIM * m;
    for (int i = tid; i < total; i += 256)
      vbuf[i] = (i < avail) ? src[i] : 0.f;
  }
  __syncthreads();

  // ---- Stage 1: Gram matrix + sign/eps clamp ----
  for (int i = tid; i < TILE_M * NI; i += 256) {
    int s = i >> 10, rem = i & 1023, a = rem >> 5, bb = rem & 31;
    const float* va = &vbuf[(s * C_DIM + a) * m];
    const float* vb = &vbuf[(s * C_DIM + bb) * m];
    float d = 0.f;
    for (int c = 0; c < m; ++c) d += va[c] * vb[c];
    if (d != 0.f) d = copysignf(fmaxf(fabsf(d), 1e-12f), d);  // sign(0)*... = 0
    Xf32[i] = d;
  }
  __syncthreads();

  // ---- Stage 2: LN1 + LeakyReLU -> bf16 ----
  ln_lrelu_stage(Xf32, Xb, g1, b1, wid, lane);
  __syncthreads();

  // ---- Stage 3: GEMM1 (WMMA bf16) ----
  gemm_1024(Xb, Xf32, W1, nullptr, wid, l15, hf);
  __syncthreads();

  // ---- Stage 4: LN2 + LeakyReLU -> bf16 ----
  ln_lrelu_stage(Xf32, Xb, g2, b2, wid, lane);
  __syncthreads();

  // ---- Stage 5: GEMM2 + bias (WMMA bf16) ----
  gemm_1024(Xb, Xf32, W2, bias2, wid, l15, hf);
  __syncthreads();

  // ---- Stage 6: softmax over groups of 32, then attn @ v ----
  for (int j = 0; j < 4; ++j) {
    int r  = tid + 256 * j;          // 0..1023 : 32 samples x 32 attn rows
    int s  = r >> 5, a = r & 31;
    int sg = s0 + s;
    if (sg >= Ntot) continue;
    const float* hrow = Xf32 + s * NI + a * C_DIM;
    float mx = hrow[0];
#pragma unroll
    for (int b = 1; b < 32; ++b) mx = fmaxf(mx, hrow[b]);
    float e[32];
    float sum = 0.f;
#pragma unroll
    for (int b = 0; b < 32; ++b) { float t = __expf(hrow[b] - mx); e[b] = t; sum += t; }
    float inv = 1.f / sum;
    for (int c = 0; c < m; ++c) {
      float o = 0.f;
#pragma unroll
      for (int b = 0; b < 32; ++b) o += e[b] * vbuf[(s * C_DIM + b) * m + c];
      out[((size_t)sg * C_DIM + a) * m + c] = o * inv;
    }
  }
}

extern "C" void kernel_launch(void* const* d_in, const int* in_sizes, int n_in,
                              void* d_out, int out_size, void* d_ws, size_t ws_size,
                              hipStream_t stream) {
  (void)n_in; (void)out_size; (void)ws_size;
  const float* f0      = (const float*)d_in[0];
  const float* f1      = (const float*)d_in[1];
  const float* ln1_g_0 = (const float*)d_in[2];
  const float* ln1_b_0 = (const float*)d_in[3];
  const float* w1_0    = (const float*)d_in[4];
  const float* ln2_g_0 = (const float*)d_in[5];
  const float* ln2_b_0 = (const float*)d_in[6];
  const float* w2_0    = (const float*)d_in[7];
  const float* b2_0    = (const float*)d_in[8];
  const float* ln1_g_1 = (const float*)d_in[9];
  const float* ln1_b_1 = (const float*)d_in[10];
  const float* w1_1    = (const float*)d_in[11];
  const float* ln2_g_1 = (const float*)d_in[12];
  const float* ln2_b_1 = (const float*)d_in[13];
  const float* w2_1    = (const float*)d_in[14];
  const float* b2_1    = (const float*)d_in[15];

  const int N = in_sizes[0] / C_DIM;   // f0 is [N, 32, 1]

  // bf16 weight staging in workspace: 4 x 1024x1024 x 2B = 8 MiB
  unsigned short* w1b_0 = (unsigned short*)d_ws;
  unsigned short* w2b_0 = w1b_0 + (size_t)NI * NI;
  unsigned short* w1b_1 = w2b_0 + (size_t)NI * NI;
  unsigned short* w2b_1 = w1b_1 + (size_t)NI * NI;

  const int nw = NI * NI;
  dim3 cgrid((nw + 255) / 256);
  convert_f32_to_bf16<<<cgrid, 256, 0, stream>>>(w1_0, w1b_0, nw);
  convert_f32_to_bf16<<<cgrid, 256, 0, stream>>>(w2_0, w2b_0, nw);
  convert_f32_to_bf16<<<cgrid, 256, 0, stream>>>(w1_1, w1b_1, nw);
  convert_f32_to_bf16<<<cgrid, 256, 0, stream>>>(w2_1, w2b_1, nw);

  const size_t shmem = (size_t)TILE_M * NI * 4   // Xf32
                     + (size_t)TILE_M * NI * 2   // Xbf16
                     + (size_t)TILE_M * C_DIM * 3 * 4;  // vbuf
  (void)hipFuncSetAttribute((const void*)gal_kernel,
                            hipFuncAttributeMaxDynamicSharedMemorySize,
                            (int)shmem);

  const int blocks = (N + TILE_M - 1) / TILE_M;
  gal_kernel<<<blocks, 256, shmem, stream>>>(
      f0, w1b_0, w2b_0, ln1_g_0, ln1_b_0, ln2_g_0, ln2_b_0, b2_0,
      (float*)d_out, 1, N);
  gal_kernel<<<blocks, 256, shmem, stream>>>(
      f1, w1b_1, w2b_1, ln1_g_1, ln1_b_1, ln2_g_1, ln2_b_1, b2_1,
      (float*)d_out + (size_t)N * C_DIM, 3, N);
}
